// PAMvidx4_51642686767658
// MI455X (gfx1250) — matrix operand order repaired and checked by reading, compiled, not run on hardware
//
#include <hip/hip_runtime.h>
#include <hip/hip_bf16.h>

typedef __attribute__((ext_vector_type(16))) _Float16 v16h;
typedef __attribute__((ext_vector_type(8)))  _Float16 v8h;
typedef __attribute__((ext_vector_type(8)))  float    v8f;

union AB16 { v16h v; v8h h[2]; };

// ---------------------------------------------------------------------------
// 3x3 conv, C=32 -> 32, pad=1, NCHW fp32 in/out, f16 WMMA compute.
// MODE 0: out = leaky_relu(conv(in), 0.1)
// MODE 1: out = conv(in) + res
// Block = 128 threads (4 waves). Each wave: 2 M-tiles (32 pixels) x 32 out-ch,
// so B fragments are loaded once per tap and reused -> 2 ds loads per WMMA.
// Block covers 128 consecutive x of one output row. grid = (H * W/128, batch).
// Weights are staged into LDS via the CDNA5 async-to-LDS DMA path (ASYNCcnt).
// ---------------------------------------------------------------------------
template <int MODE>
__global__ __launch_bounds__(128)
void conv3x3_wmma(const float* __restrict__ in, const float* __restrict__ wgt,
                  const float* __restrict__ res, float* __restrict__ out,
                  int Hh, int Ww) {
  __shared__ __attribute__((aligned(16))) _Float16 sIn[3 * 130 * 32]; // [dy][xx][c]
  __shared__ __attribute__((aligned(16))) _Float16 sW[9 * 32 * 32];   // [t][n][c]
  __shared__ __attribute__((aligned(16))) float sWstage[9 * 32 * 32]; // raw [n][c][t]

  const int tid = threadIdx.x;              // 0..127
  const int xblocks = Ww >> 7;
  const int y0 = blockIdx.x / xblocks;
  const int xblk = (blockIdx.x % xblocks) << 7;
  const int bz = blockIdx.y;
  const size_t plane = (size_t)Hh * (size_t)Ww;
  const float* inb = in + (size_t)bz * 32 * plane;
  float* outb = out + (size_t)bz * 32 * plane;

  // --- async DMA: raw fp32 weights -> LDS staging (9216 floats, 2304 b128) ---
  for (int i = tid * 4; i < 9 * 32 * 32; i += 128 * 4) {
    unsigned ldsa = (unsigned)(unsigned long long)(void*)&sWstage[i];
    asm volatile("global_load_async_to_lds_b128 %0, %1, off"
                 :: "v"(ldsa), "v"(wgt + i) : "memory");
  }

  // --- input tile on the normal path (needs f32->f16 cvt + zero padding) ---
  // rows y0-1..y0+1, x = xblk-1 .. xblk+128; i = (dy*32 + c)*130 + xx
  for (int i = tid; i < 3 * 32 * 130; i += 128) {
    int xx = i % 130;
    int t2 = i / 130;
    int c = t2 & 31;
    int dy = t2 >> 5;
    int gy = y0 - 1 + dy;
    int gx = xblk - 1 + xx;
    float v = 0.f;
    if ((unsigned)gy < (unsigned)Hh && (unsigned)gx < (unsigned)Ww)
      v = inb[(size_t)c * plane + (size_t)gy * Ww + gx];
    sIn[(dy * 130 + xx) * 32 + c] = (_Float16)v;
  }

  asm volatile("s_wait_asynccnt 0x0" ::: "memory");
  __syncthreads();

  // --- convert + transpose weights: sW[(t*32+n)*32+c] = w[n][c][t] ---
  for (int i = tid; i < 9 * 32 * 32; i += 128) {
    int t = i >> 10;
    int rem = i & 1023;
    int n = rem >> 5;
    int c = rem & 31;
    sW[i] = (_Float16)sWstage[(n * 32 + c) * 9 + t];
  }
  __syncthreads();

  const int lane = tid & 31;
  const int wv = tid >> 5;     // 0..3
  const int half = lane >> 4;  // K-half selector for A, B
  const int m = lane & 15;     // A matrix row (pixel within M-tile)
  const int nlo = lane & 15;   // D/B column
  const int x0 = wv << 5;      // wave's 32-pixel base within the 128-px block

  v8f acc[2][2];               // [mtile][ntile]
#pragma unroll
  for (int i = 0; i < 2; ++i)
#pragma unroll
    for (int j = 0; j < 2; ++j)
      acc[i][j] = (v8f){0.f, 0.f, 0.f, 0.f, 0.f, 0.f, 0.f, 0.f};

#pragma unroll
  for (int ry = 0; ry < 3; ++ry) {
#pragma unroll
    for (int rx = 0; rx < 3; ++rx) {
      const int t = ry * 3 + rx;
      // B fragments (shared by both M-tiles): lanes 0-15 K=0..15, 16-31 K=16..31
      AB16 b0, b1;
      {
        const _Float16* pb = &sW[(t * 32 + nlo) * 32 + half * 16];
        b0.h[0] = *(const v8h*)(pb);
        b0.h[1] = *(const v8h*)(pb + 8);
      }
      {
        const _Float16* pb = &sW[(t * 32 + 16 + nlo) * 32 + half * 16];
        b1.h[0] = *(const v8h*)(pb);
        b1.h[1] = *(const v8h*)(pb + 8);
      }
#pragma unroll
      for (int mt = 0; mt < 2; ++mt) {
        // A fragment: lane half h -> K in h*16..h*16+15
        const _Float16* pa = &sIn[((ry * 130) + x0 + mt * 16 + m + rx) * 32];
        AB16 a;
        a.h[0] = *(const v8h*)(pa + half * 16);
        a.h[1] = *(const v8h*)(pa + half * 16 + 8);
        acc[mt][0] = __builtin_amdgcn_wmma_f32_16x16x32_f16(
            false, a.v, false, b0.v, (short)0, acc[mt][0], false, false);
        acc[mt][1] = __builtin_amdgcn_wmma_f32_16x16x32_f16(
            false, a.v, false, b1.v, (short)0, acc[mt][1], false, false);
      }
    }
  }

  // D layout: lane column n = lane&15; VGPR v -> M row = v + half*8 (pixel x).
#pragma unroll
  for (int mt = 0; mt < 2; ++mt) {
    const int xout = xblk + x0 + mt * 16 + half * 8;
#pragma unroll
    for (int nt = 0; nt < 2; ++nt) {
      const int n = nt * 16 + nlo;
      float* po = outb + ((size_t)n * Hh + y0) * Ww + xout;
      const float* pr = (MODE == 1)
          ? res + (size_t)bz * 32 * plane + ((size_t)n * Hh + y0) * Ww + xout
          : nullptr;
#pragma unroll
      for (int v = 0; v < 8; ++v) {
        float val = acc[mt][nt][v];
        if (MODE == 0) val = val > 0.f ? val : 0.1f * val;
        else val += pr[v];
        po[v] = val;
      }
    }
  }
}

// ---------------------------------------------------------------------------
// 1x1 conv, C=32 -> 16, + bias. One WMMA per wave (16 pixels).
// grid = (npix/128, batch), block 256.
// ---------------------------------------------------------------------------
__global__ __launch_bounds__(256)
void conv1x1_n16(const float* __restrict__ in, const float* __restrict__ wgt,
                 const float* __restrict__ bias, float* __restrict__ out,
                 int npix) {
  const int tid = threadIdx.x;
  const int lane = tid & 31;
  const int wv = tid >> 5;
  const int half = lane >> 4;
  const int m = lane & 15;
  const int nlo = lane & 15;
  const int bz = blockIdx.y;
  const int p0 = blockIdx.x * 128 + wv * 16;
  const int p = p0 + m;
  const float* inb = in + (size_t)bz * 32 * npix;
  float* outb = out + (size_t)bz * 16 * npix;

  AB16 a;
#pragma unroll
  for (int j = 0; j < 8; ++j) {
    const int k0 = half * 8 + j;
    const int k1 = 16 + half * 8 + j;
    a.h[0][j] = (_Float16)inb[(size_t)k0 * npix + p];
    a.h[1][j] = (_Float16)inb[(size_t)k1 * npix + p];
  }
  AB16 b;
#pragma unroll
  for (int e = 0; e < 8; ++e) {
    b.h[0][e] = (_Float16)wgt[nlo * 32 + half * 8 + e];
    b.h[1][e] = (_Float16)wgt[nlo * 32 + 16 + half * 8 + e];
  }
  v8f acc = {0.f, 0.f, 0.f, 0.f, 0.f, 0.f, 0.f, 0.f};
  acc = __builtin_amdgcn_wmma_f32_16x16x32_f16(
      false, a.v, false, b.v, (short)0, acc, false, false);

  const float bs = bias[nlo];
  float* po = outb + (size_t)nlo * npix + p0 + half * 8;
#pragma unroll
  for (int v = 0; v < 8; ++v) po[v] = acc[v] + bs;
}

// ---------------------------------------------------------------------------
// fepam: per-query gather attention. One wave per query.
// Q: (b,16,h,w)  S: (b,16,H,W)  R: (b,32,H,W)  px,py: (b,h*w,64)
// out: (b,32,h,w). k=64: lane handles k=lane and k=lane+32 for the scores,
// then lane = value channel v for the output accumulation (attn/idx via shfl).
// S/R working sets (<= 96 MB) are L2-resident (192 MB), so gathers hit L2.
// ---------------------------------------------------------------------------
__global__ __launch_bounds__(256)
void fepam_kernel(const float* __restrict__ Q, const float* __restrict__ S,
                  const float* __restrict__ R, const int* __restrict__ px,
                  const int* __restrict__ py, float* __restrict__ out,
                  int hw, int H, int W) {
  const int lane = threadIdx.x & 31;
  const int wv = threadIdx.x >> 5;
  const int q = blockIdx.x * 8 + wv;
  const int bz = blockIdx.y;
  const size_t HW = (size_t)H * (size_t)W;

  const float* Qb = Q + (size_t)bz * 16 * hw;
  const float* Sb = S + (size_t)bz * 16 * HW;
  const float* Rb = R + (size_t)bz * 32 * HW;
  const int* pxb = px + ((size_t)bz * hw + q) * 64;
  const int* pyb = py + ((size_t)bz * hw + q) * 64;

  float Qv[16];
#pragma unroll
  for (int c = 0; c < 16; ++c) Qv[c] = Qb[(size_t)c * hw + q];

  const int i0 = pxb[lane] * W + pyb[lane];
  const int i1 = pxb[lane + 32] * W + pyb[lane + 32];

  float s0 = 0.f, s1 = 0.f;
#pragma unroll
  for (int c = 0; c < 16; ++c) {
    const float* Sc = Sb + (size_t)c * HW;
    s0 += Qv[c] * Sc[i0];
    s1 += Qv[c] * Sc[i1];
  }
  // softmax over the 64 scores of this wave
  float mx = fmaxf(s0, s1);
#pragma unroll
  for (int off = 16; off > 0; off >>= 1) mx = fmaxf(mx, __shfl_xor(mx, off));
  const float e0 = __expf(s0 - mx);
  const float e1 = __expf(s1 - mx);
  float sum = e0 + e1;
#pragma unroll
  for (int off = 16; off > 0; off >>= 1) sum += __shfl_xor(sum, off);
  const float inv = 1.f / sum;
  const float a0 = e0 * inv;
  const float a1 = e1 * inv;

  // lane = value channel (32 lanes = vC 32)
  const float* Rv = Rb + (size_t)lane * HW;
  float acc = 0.f;
#pragma unroll 4
  for (int k = 0; k < 32; ++k) {
    const float ak = __shfl(a0, k);
    const int ik = __shfl(i0, k);
    acc += ak * Rv[ik];
  }
#pragma unroll 4
  for (int k = 0; k < 32; ++k) {
    const float ak = __shfl(a1, k);
    const int ik = __shfl(i1, k);
    acc += ak * Rv[ik];
  }
  out[(size_t)bz * 32 * hw + (size_t)lane * hw + q] = acc;
}

// ---------------------------------------------------------------------------
// Final fused 1x1 conv: concat[b1,b2,b3,lb0] (K=128) -> 32 channels + bias.
// 8 WMMAs per wave (4 K-chunks x 2 N-tiles). grid = (npix/128, batch).
// ---------------------------------------------------------------------------
__global__ __launch_bounds__(256)
void fuse1x1(const float* __restrict__ s0p, const float* __restrict__ s1p,
             const float* __restrict__ s2p, const float* __restrict__ s3p,
             const float* __restrict__ wf, const float* __restrict__ bf,
             float* __restrict__ out, int npix) {
  const int tid = threadIdx.x;
  const int lane = tid & 31;
  const int wv = tid >> 5;
  const int half = lane >> 4;
  const int m = lane & 15;
  const int nlo = lane & 15;
  const int bz = blockIdx.y;
  const int p0 = blockIdx.x * 128 + wv * 16;
  const int p = p0 + m;

  const float* srcs[4] = {s0p + (size_t)bz * 32 * npix, s1p + (size_t)bz * 32 * npix,
                          s2p + (size_t)bz * 32 * npix, s3p + (size_t)bz * 32 * npix};

  v8f acc0 = {0.f, 0.f, 0.f, 0.f, 0.f, 0.f, 0.f, 0.f};
  v8f acc1 = {0.f, 0.f, 0.f, 0.f, 0.f, 0.f, 0.f, 0.f};

#pragma unroll
  for (int s = 0; s < 4; ++s) {
    const float* sb = srcs[s];
    AB16 a;
#pragma unroll
    for (int j = 0; j < 8; ++j) {
      const int k0 = half * 8 + j;
      const int k1 = 16 + half * 8 + j;
      a.h[0][j] = (_Float16)sb[(size_t)k0 * npix + p];
      a.h[1][j] = (_Float16)sb[(size_t)k1 * npix + p];
    }
    {
      AB16 b;
#pragma unroll
      for (int e = 0; e < 8; ++e) {
        b.h[0][e] = (_Float16)wf[nlo * 128 + s * 32 + half * 8 + e];
        b.h[1][e] = (_Float16)wf[nlo * 128 + s * 32 + 16 + half * 8 + e];
      }
      acc0 = __builtin_amdgcn_wmma_f32_16x16x32_f16(
          false, a.v, false, b.v, (short)0, acc0, false, false);
    }
    {
      AB16 b;
#pragma unroll
      for (int e = 0; e < 8; ++e) {
        b.h[0][e] = (_Float16)wf[(16 + nlo) * 128 + s * 32 + half * 8 + e];
        b.h[1][e] = (_Float16)wf[(16 + nlo) * 128 + s * 32 + 16 + half * 8 + e];
      }
      acc1 = __builtin_amdgcn_wmma_f32_16x16x32_f16(
          false, a.v, false, b.v, (short)0, acc1, false, false);
    }
  }

  float* ob = out + (size_t)bz * 32 * npix;
  {
    const float bs = bf[nlo];
    float* po = ob + (size_t)nlo * npix + p0 + half * 8;
#pragma unroll
    for (int v = 0; v < 8; ++v) po[v] = acc0[v] + bs;
  }
  {
    const float bs = bf[16 + nlo];
    float* po = ob + (size_t)(16 + nlo) * npix + p0 + half * 8;
#pragma unroll
    for (int v = 0; v < 8; ++v) po[v] = acc1[v] + bs;
  }
}

// ---------------------------------------------------------------------------
// Host-side orchestration
// ---------------------------------------------------------------------------
extern "C" void kernel_launch(void* const* d_in, const int* in_sizes, int n_in,
                              void* d_out, int out_size, void* d_ws, size_t ws_size,
                              hipStream_t stream) {
  const float* lr0 = (const float*)d_in[0];
  const float* lr1 = (const float*)d_in[1];
  const float* lr2 = (const float*)d_in[2];
  const float* hr  = (const float*)d_in[3];
  const float* wl1 = (const float*)d_in[4];
  const float* wl2 = (const float*)d_in[5];
  const float* wh1 = (const float*)d_in[6];
  const float* wh2 = (const float*)d_in[7];
  const float* wq  = (const float*)d_in[8];
  const float* bq  = (const float*)d_in[9];
  const float* wk  = (const float*)d_in[10];
  const float* bk  = (const float*)d_in[11];
  const float* whk = (const float*)d_in[12];
  const float* bhk = (const float*)d_in[13];
  const float* wf  = (const float*)d_in[14];
  const float* bf  = (const float*)d_in[15];
  const int* px0 = (const int*)d_in[16];
  const int* py0 = (const int*)d_in[17];
  const int* px1 = (const int*)d_in[18];
  const int* py1 = (const int*)d_in[19];
  const int* px2 = (const int*)d_in[20];
  const int* py2 = (const int*)d_in[21];
  float* dout = (float*)d_out;

  // Workspace layout (floats). Total 36,700,160 floats = 140 MiB.
  // Q/S1/S2/S3/b1-b3 alias the resb TMP region (dead after the resbs).
  float* ws = (float*)d_ws;
  const size_t M1 = 1048576;            // per LR tensor = 1M floats
  float* LB0 = ws;                      // (2,32,128,128)
  float* LB1 = ws + M1;
  float* LB2 = ws + 2 * M1;
  float* HB  = ws + 3 * M1;             // (2,32,512,512) = 16M floats
  float* TMP = ws + 3 * M1 + 16777216;  // 16M floats, resb intermediate
  float* Qb = TMP;                      // (2,16,128,128) = 512K floats
  float* S1 = Qb + 524288;
  float* S2 = S1 + 524288;
  float* S3 = S2 + 524288;              // (2,16,512,512) = 8M floats
  float* B1 = S3 + 8388608;             // (2,32,128,128) = 1M floats
  float* B2 = B1 + M1;
  float* B3 = B2 + M1;
  (void)in_sizes; (void)n_in; (void)out_size; (void)ws_size;

  const dim3 blk128(128);
  const dim3 blk256(256);
  // LR residual blocks (shared weights wl1/wl2), TMP reused sequentially.
  conv3x3_wmma<0><<<dim3(128, 2), blk128, 0, stream>>>(lr0, wl1, nullptr, TMP, 128, 128);
  conv3x3_wmma<1><<<dim3(128, 2), blk128, 0, stream>>>(TMP, wl2, lr0, LB0, 128, 128);
  conv3x3_wmma<0><<<dim3(128, 2), blk128, 0, stream>>>(lr1, wl1, nullptr, TMP, 128, 128);
  conv3x3_wmma<1><<<dim3(128, 2), blk128, 0, stream>>>(TMP, wl2, lr1, LB1, 128, 128);
  conv3x3_wmma<0><<<dim3(128, 2), blk128, 0, stream>>>(lr2, wl1, nullptr, TMP, 128, 128);
  conv3x3_wmma<1><<<dim3(128, 2), blk128, 0, stream>>>(TMP, wl2, lr2, LB2, 128, 128);
  // HR residual block.
  conv3x3_wmma<0><<<dim3(2048, 2), blk128, 0, stream>>>(hr, wh1, nullptr, TMP, 512, 512);
  conv3x3_wmma<1><<<dim3(2048, 2), blk128, 0, stream>>>(TMP, wh2, hr, HB, 512, 512);
  // 1x1 projections (Q, S1, S2, S3).
  conv1x1_n16<<<dim3(128, 2), blk256, 0, stream>>>(LB0, wq, bq, Qb, 16384);
  conv1x1_n16<<<dim3(128, 2), blk256, 0, stream>>>(LB1, wk, bk, S1, 16384);
  conv1x1_n16<<<dim3(128, 2), blk256, 0, stream>>>(LB2, wk, bk, S2, 16384);
  conv1x1_n16<<<dim3(2048, 2), blk256, 0, stream>>>(HB, whk, bhk, S3, 262144);
  // Gather attention (one wave per query; 2048*8 = 16384 queries per batch).
  fepam_kernel<<<dim3(2048, 2), blk256, 0, stream>>>(Qb, S1, lr1, px0, py0, B1, 16384, 128, 128);
  fepam_kernel<<<dim3(2048, 2), blk256, 0, stream>>>(Qb, S2, lr2, px1, py1, B2, 16384, 128, 128);
  fepam_kernel<<<dim3(2048, 2), blk256, 0, stream>>>(Qb, S3, hr, px2, py2, B3, 16384, 512, 512);
  // Final fused 1x1 over concat[b1,b2,b3,lb0].
  fuse1x1<<<dim3(128, 2), blk256, 0, stream>>>(B1, B2, B3, LB0, wf, bf, dout, 16384);
}
